// VisualQuery_49160195670429
// MI455X (gfx1250) — matrix operand order repaired and checked by reading, not compiled, not run on hardware
//
#include <hip/hip_runtime.h>
#include <hip/hip_bf16.h>

typedef _Float16 v16h __attribute__((ext_vector_type(16)));
typedef _Float16 v8h  __attribute__((ext_vector_type(8)));
typedef float    v8f  __attribute__((ext_vector_type(8)));

/* global (AS1) view of a B fragment: guarantees global_load_b128 lowering */
typedef __attribute__((address_space(1))) const v16h gv16h_c;

#define B_   8
#define H_   64
#define W_   64
#define C_   64
#define N_   16384
#define P_   (B_ * N_)      /* 131072 points */
#define HID_ 256
#define K0P_ 96             /* layer0 K padded 66 -> 96 */
#define TILE_M 64           /* points per workgroup */
#define THREADS 512         /* 16 wave32 per block; 4 col-tiles per wave */

/* -------- workspace layout (bytes); total ~98.7 MB -------- */
static constexpr size_t OFF_X   = 0;                                  // [4][P][96] f16
static constexpr size_t OFF_WGT = OFF_X   + (size_t)4*P_*K0P_*2;      // [4][P] f32 (normalized area weights)
static constexpr size_t OFF_W0P = OFF_WGT + (size_t)4*P_*4;           // packed w0 [96][256] f16
static constexpr size_t OFF_W1P = OFF_W0P + (size_t)K0P_*256*2;
static constexpr size_t OFF_W2P = OFF_W1P + (size_t)256*256*2;
static constexpr size_t OFF_W3P = OFF_W2P + (size_t)256*256*2;
static constexpr size_t OFF_W4P = OFF_W3P + (size_t)256*256*2;
static constexpr size_t OFF_WOP = OFF_W4P + (size_t)256*256*2;

/* ---------------------------------------------------------------------------
 * pack_w_kernel: pre-swizzle weights [K][256] f32 row-major into the B-matrix
 * fragment layout of V_WMMA_F32_16X16X32_F16 (ISA 7.12.2):
 *   fragment(nt, kt): lane l holds 16 f16 = W[kt*32 + ((l&16)?8:0) + i + (i>=8?8:0)][nt*16 + (l&15)]
 * stored contiguously so each lane loads its fragment with one 32B load.
 * ------------------------------------------------------------------------- */
__global__ void pack_w_kernel(const float* __restrict__ src, _Float16* __restrict__ dst,
                              int Ksrc, int kTiles) {
    int t = blockIdx.x * blockDim.x + threadIdx.x;
    int total = 16 * kTiles * 32;
    if (t >= total) return;
    int lane  = t & 31;
    int kt    = (t >> 5) % kTiles;
    int nt    = (t >> 5) / kTiles;
    int n     = nt * 16 + (lane & 15);
    int kbase = kt * 32 + ((lane & 16) ? 8 : 0);
#pragma unroll
    for (int i = 0; i < 16; ++i) {
        int kk = kbase + i + ((i >= 8) ? 8 : 0);
        float v = (kk < Ksrc) ? src[(size_t)kk * 256 + n] : 0.0f;
        dst[(size_t)t * 16 + i] = (_Float16)v;
    }
}

/* ---------------------------------------------------------------------------
 * prep_kernel: nearest-gather + rel coords + area weights (reference math).
 * ------------------------------------------------------------------------- */
__device__ __forceinline__ int nearest_idx(float c, float nf, int nmax) {
    float x = rintf(((c + 1.0f) * nf - 1.0f) * 0.5f);   /* v_rndne: half-to-even */
    int i = (int)x;
    return i < 0 ? 0 : (i > nmax ? nmax : i);
}

__global__ void prep_kernel(const float* __restrict__ feat, const float* __restrict__ coord,
                            _Float16* __restrict__ Xg, float* __restrict__ wgt) {
    int p = blockIdx.x * blockDim.x + threadIdx.x;
    if (p >= P_) return;
    int b = p >> 14;                                    /* p / N_ */
    float c0 = coord[2 * p + 0];
    float c1 = coord[2 * p + 1];
    const float r = 1.0f / 64.0f;
    float area[4];
#pragma unroll
    for (int k = 0; k < 4; ++k) {
        float vx = (k & 2) ? 1.0f : -1.0f;              /* vx outer, vy inner in reference */
        float vy = (k & 1) ? 1.0f : -1.0f;
        float cx = fminf(fmaxf(c0 + vx * r + 1e-6f, -1.0f + 1e-6f), 1.0f - 1e-6f);
        float cy = fminf(fmaxf(c1 + vy * r + 1e-6f, -1.0f + 1e-6f), 1.0f - 1e-6f);
        int ix = nearest_idx(cx, 64.0f, W_ - 1);
        int iy = nearest_idx(cy, 64.0f, H_ - 1);
        float qc0 = -1.0f + (2.0f * (float)iy + 1.0f) / 64.0f;
        float qc1 = -1.0f + (2.0f * (float)ix + 1.0f) / 64.0f;
        float rel0 = (c0 - qc0) * 64.0f;
        float rel1 = (c1 - qc1) * 64.0f;
        area[k] = 1.0f / (fabsf(rel0 * rel1) + 1e-9f);

        const float4* frow = (const float4*)(feat + (((size_t)b * H_ + iy) * W_ + ix) * C_);
        _Float16* xr = Xg + ((size_t)k * P_ + p) * K0P_;
#pragma unroll
        for (int j = 0; j < 16; ++j) {
            float4 f = frow[j];
            xr[4 * j + 0] = (_Float16)f.x;
            xr[4 * j + 1] = (_Float16)f.y;
            xr[4 * j + 2] = (_Float16)f.z;
            xr[4 * j + 3] = (_Float16)f.w;
        }
        xr[64] = (_Float16)rel0;
        xr[65] = (_Float16)rel1;
#pragma unroll
        for (int j = 66; j < 96; ++j) xr[j] = (_Float16)0.0f;
    }
    float tot = area[0] + area[1] + area[2] + area[3];
#pragma unroll
    for (int k = 0; k < 4; ++k) wgt[(size_t)k * P_ + p] = area[k] / tot;
}

/* ---------------------------------------------------------------------------
 * Quad-tile GEMM: one wave computes 4 adjacent 16x16 output tiles (same 16
 * rows). A fragment (ds_load_b128 x2) shared by all 4 tiles per K-step;
 * 4 independent v_wmma accumulator chains per K-step for ILP. B fragments
 * stream from the pre-swizzled L2-resident weights via global_load_b128.
 * K loop kept rolled (unroll 1) to bound live fragments (no spills).
 * ------------------------------------------------------------------------- */
__device__ __forceinline__ void gemm_quad(const _Float16* src, int kTiles,
                                          const _Float16* __restrict__ wp,
                                          const float* __restrict__ bias,
                                          int lane, int mt, int ntb, v8f (&acc)[4]) {
    const int m  = mt * 16 + (lane & 15);
    const int kb = (lane & 16) ? 8 : 0;
#pragma unroll
    for (int t = 0; t < 4; ++t) {
        float bv = bias[(ntb + t) * 16 + (lane & 15)];
#pragma unroll
        for (int i = 0; i < 8; ++i) acc[t][i] = bv;
    }
#pragma unroll 1
    for (int kt = 0; kt < kTiles; ++kt) {
        const _Float16* arow = src + m * 256 + kt * 32 + kb;
        union { v16h v; v8h h[2]; } ua;
        ua.h[0] = *(const v8h*)(arow);
        ua.h[1] = *(const v8h*)(arow + 16);
        int ktn = (kt + 1 < kTiles) ? kt + 1 : kt;       /* branchless clamp */
        __builtin_prefetch(wp + ((size_t)(ntb * kTiles + ktn) * 32 + lane) * 16, 0, 3);
#pragma unroll
        for (int t = 0; t < 4; ++t) {
            v16h bf = *(gv16h_c*)(wp + ((size_t)((ntb + t) * kTiles + kt) * 32 + lane) * 16);
            acc[t] = __builtin_amdgcn_wmma_f32_16x16x32_f16(false, ua.v, false, bf,
                                                            (short)0, acc[t], false, false);
        }
    }
}

/* Scatter D tile (f32) to LDS as f16: lane<16 rows v, lane>=16 rows 8+v */
__device__ __forceinline__ void store_tile(v8f acc, _Float16* dst,
                                           int lane, int mt, int nt, bool relu) {
    const int col = nt * 16 + (lane & 15);
    const int rb  = mt * 16 + ((lane & 16) ? 8 : 0);
#pragma unroll
    for (int v = 0; v < 8; ++v) {
        float f = acc[v];
        if (relu) f = (f < 0.0f) ? 0.0f : f;
        dst[(rb + v) * 256 + col] = (_Float16)f;
    }
}

__global__ __launch_bounds__(THREADS) void mlp_kernel(
    const _Float16* __restrict__ Xg, const float* __restrict__ wgtp,
    const _Float16* __restrict__ w0p, const _Float16* __restrict__ w1p,
    const _Float16* __restrict__ w2p, const _Float16* __restrict__ w3p,
    const _Float16* __restrict__ w4p, const _Float16* __restrict__ wop,
    const float* __restrict__ b0, const float* __restrict__ b1,
    const float* __restrict__ b2, const float* __restrict__ b3,
    const float* __restrict__ b4, const float* __restrict__ bo,
    const float* __restrict__ lng, const float* __restrict__ lnb,
    float* __restrict__ out) {
    __shared__ __align__(16) unsigned char smem[65536];
    _Float16* Xa = (_Float16*)smem;            /* [64][256] f16 */
    _Float16* Hb = (_Float16*)(smem + 32768);  /* [64][256] f16 */
    float*    Yf = (float*)smem;               /* [64][256] f32 overlay (final stage) */

    const int tid  = threadIdx.x;
    const int lane = tid & 31;
    const int wave = tid >> 5;                 /* 0..15 */
    const int mt   = wave & 3;                 /* row tile: rows mt*16..mt*16+15 */
    const int ntb  = (wave >> 2) * 4;          /* 4 column tiles per wave */
    const int base = blockIdx.x * TILE_M;
    const int rb   = mt * 16 + ((lane & 16) ? 8 : 0);

    v8f accOut[4];                             /* blended pre-wo activations, f32 */
#pragma unroll
    for (int t = 0; t < 4; ++t)
#pragma unroll
        for (int i = 0; i < 8; ++i) accOut[t][i] = 0.0f;

    /* Opaque zero offset, redefined every corner: blocks LICM from hoisting
     * (and spilling) the loop-invariant B-fragment loads out of the corner
     * loop, while preserving pointer provenance (loads stay global_*). */
    size_t kOff = 0;

    for (int k = 0; k < 4; ++k) {
        asm volatile("" : "+s"(kOff));
        const _Float16* w0l = w0p + kOff;
        const _Float16* w1l = w1p + kOff;
        const _Float16* w2l = w2p + kOff;
        const _Float16* w3l = w3p + kOff;
        const _Float16* w4l = w4p + kOff;

        /* stage this corner's inputs: 64 rows x 96 halves -> Xa (stride 256) */
        for (int idx = tid; idx < 64 * 12; idx += THREADS) {
            int r = idx / 12, c4 = idx % 12;
            const uint4* src = (const uint4*)(Xg + ((size_t)k * P_ + base + r) * K0P_);
            ((uint4*)(Xa + r * 256))[c4] = src[c4];
        }
        __syncthreads();

        v8f a[4];
        gemm_quad(Xa, 3, w0l, b0, lane, mt, ntb, a);            /* L0: K=96 */
#pragma unroll
        for (int t = 0; t < 4; ++t) store_tile(a[t], Hb, lane, mt, ntb + t, true);
        __syncthreads();

        gemm_quad(Hb, 8, w1l, b1, lane, mt, ntb, a);            /* L1 */
#pragma unroll
        for (int t = 0; t < 4; ++t) store_tile(a[t], Xa, lane, mt, ntb + t, true);
        __syncthreads();

        gemm_quad(Xa, 8, w2l, b2, lane, mt, ntb, a);            /* L2 */
#pragma unroll
        for (int t = 0; t < 4; ++t) store_tile(a[t], Hb, lane, mt, ntb + t, true);
        __syncthreads();

        gemm_quad(Hb, 8, w3l, b3, lane, mt, ntb, a);            /* L3 */
#pragma unroll
        for (int t = 0; t < 4; ++t) store_tile(a[t], Xa, lane, mt, ntb + t, true);
        __syncthreads();

        gemm_quad(Xa, 8, w4l, b4, lane, mt, ntb, a);            /* L4 + blend */
#pragma unroll
        for (int t = 0; t < 4; ++t)
#pragma unroll
            for (int v = 0; v < 8; ++v) {
                float s = wgtp[(size_t)k * P_ + base + rb + v];
                accOut[t][v] += s * a[t][v];
            }
        __syncthreads();                       /* Xa free for next corner */
    }

    /* final linear wo on blended activations */
#pragma unroll
    for (int t = 0; t < 4; ++t) store_tile(accOut[t], Xa, lane, mt, ntb + t, false);
    __syncthreads();
    gemm_quad(Xa, 8, wop, bo, lane, mt, ntb, accOut);
    __syncthreads();                           /* all Xa reads done -> safe to overlay Yf */
#pragma unroll
    for (int t = 0; t < 4; ++t) {
        int col = (ntb + t) * 16 + (lane & 15);
#pragma unroll
        for (int v = 0; v < 8; ++v) Yf[(rb + v) * 256 + col] = accOut[t][v];
    }
    __syncthreads();

    /* LayerNorm: one row per thread for tid<64 */
    if (tid < 64) {
        const float* yr = Yf + tid * 256;
        float s = 0.0f, ss = 0.0f;
        for (int c = 0; c < 256; ++c) { float y = yr[c]; s += y; ss += y * y; }
        float mu   = s * (1.0f / 256.0f);
        float var  = ss * (1.0f / 256.0f) - mu * mu;
        float rinv = rsqrtf(var + 1e-5f);
        float* orow = out + (size_t)(base + tid) * 256;
        for (int c = 0; c < 256; ++c)
            orow[c] = (yr[c] - mu) * rinv * lng[c] + lnb[c];
    }
}

extern "C" void kernel_launch(void* const* d_in, const int* in_sizes, int n_in,
                              void* d_out, int out_size, void* d_ws, size_t ws_size,
                              hipStream_t stream) {
    const float* feat  = (const float*)d_in[0];
    const float* coord = (const float*)d_in[1];
    const float* w0 = (const float*)d_in[2];  const float* b0 = (const float*)d_in[3];
    const float* w1 = (const float*)d_in[4];  const float* b1 = (const float*)d_in[5];
    const float* w2 = (const float*)d_in[6];  const float* b2 = (const float*)d_in[7];
    const float* w3 = (const float*)d_in[8];  const float* b3 = (const float*)d_in[9];
    const float* w4 = (const float*)d_in[10]; const float* b4 = (const float*)d_in[11];
    const float* wo = (const float*)d_in[12]; const float* bo = (const float*)d_in[13];
    const float* lng = (const float*)d_in[14]; const float* lnb = (const float*)d_in[15];
    float* out = (float*)d_out;

    unsigned char* ws = (unsigned char*)d_ws;   /* needs ~99 MB */
    _Float16* Xg  = (_Float16*)(ws + OFF_X);
    float*    wgt = (float*)(ws + OFF_WGT);
    _Float16* w0p = (_Float16*)(ws + OFF_W0P);
    _Float16* w1p = (_Float16*)(ws + OFF_W1P);
    _Float16* w2p = (_Float16*)(ws + OFF_W2P);
    _Float16* w3p = (_Float16*)(ws + OFF_W3P);
    _Float16* w4p = (_Float16*)(ws + OFF_W4P);
    _Float16* wop = (_Float16*)(ws + OFF_WOP);

    pack_w_kernel<<<6, 256, 0, stream>>>(w0, w0p, 66, 3);      /* 16*3*32 = 1536 thr */
    pack_w_kernel<<<16, 256, 0, stream>>>(w1, w1p, 256, 8);    /* 16*8*32 = 4096 thr */
    pack_w_kernel<<<16, 256, 0, stream>>>(w2, w2p, 256, 8);
    pack_w_kernel<<<16, 256, 0, stream>>>(w3, w3p, 256, 8);
    pack_w_kernel<<<16, 256, 0, stream>>>(w4, w4p, 256, 8);
    pack_w_kernel<<<16, 256, 0, stream>>>(wo, wop, 256, 8);

    prep_kernel<<<P_ / 256, 256, 0, stream>>>(feat, coord, Xg, wgt);

    mlp_kernel<<<P_ / TILE_M, THREADS, 0, stream>>>(Xg, wgt,
        w0p, w1p, w2p, w3p, w4p, wop,
        b0, b1, b2, b3, b4, bo, lng, lnb, out);
}
